// DynamicConv2d_37615323578786
// MI455X (gfx1250) — matrix operand order repaired and checked by reading, not compiled, not run on hardware
//
#include <hip/hip_runtime.h>
#include <hip/hip_bf16.h>
#include <math.h>

// Problem constants (from reference)
#define BB 16
#define CC 256
#define OO 256
#define EE 4
#define HH 64
#define WW 64
#define KK 3
#define KTOT (CC * KK * KK)   // 2304, weight K order: k = (kh*3+kw)*256 + c

#define APITCH 40             // LDS pitch (elems) for A tiles: 80B stride, conflict-free
#define XPITCH 40             // LDS pitch (elems) for x stage

typedef __attribute__((ext_vector_type(4)))  unsigned int u32x4;
typedef __attribute__((ext_vector_type(16))) __bf16       v16bf;
typedef __attribute__((ext_vector_type(4)))  __bf16       bf16x4;
typedef __attribute__((ext_vector_type(8)))  float        v8f;

union Frag  { v16bf v; u32x4 q[2]; };
union Pack8 { u32x4 q; __bf16 h[8]; };

// ---------------------------------------------------------------------------
// 1) Global average pool: pooled[b*C+c] = mean over H*W. One block per (b,c).
// ---------------------------------------------------------------------------
__global__ __launch_bounds__(256) void pool_kernel(const float* __restrict__ x,
                                                   float* __restrict__ pooled) {
    const int bc  = blockIdx.x;
    const int tid = threadIdx.x;
    const float* p = x + (size_t)bc * (HH * WW);
    float s = 0.f;
    #pragma unroll
    for (int i = 0; i < (HH * WW) / 256; ++i) s += p[tid + i * 256];
    __shared__ float red[256];
    red[tid] = s;
    __syncthreads();
    for (int off = 128; off > 0; off >>= 1) {
        if (tid < off) red[tid] += red[tid + off];
        __syncthreads();
    }
    if (tid == 0) pooled[bc] = red[0] * (1.0f / (HH * WW));
}

// ---------------------------------------------------------------------------
// 2) Gating: gates[b,e] = softmax_e( pooled[b,:] . gate_w[e,:] ). 64 threads.
// ---------------------------------------------------------------------------
__global__ __launch_bounds__(64) void gate_kernel(const float* __restrict__ pooled,
                                                  const float* __restrict__ gate_w,
                                                  float* __restrict__ gates) {
    const int t = threadIdx.x;      // 0..63
    const int b = t >> 2;
    const int e = t & 3;
    float s = 0.f;
    const float* pp = pooled + b * CC;
    const float* gw = gate_w + e * CC;
    #pragma unroll 8
    for (int c = 0; c < CC; ++c) s += pp[c] * gw[c];
    __shared__ float dots[BB][EE];
    __shared__ float exps[BB][EE];
    dots[b][e] = s;
    __syncthreads();
    float m = fmaxf(fmaxf(dots[b][0], dots[b][1]), fmaxf(dots[b][2], dots[b][3]));
    float ex = __expf(s - m);
    exps[b][e] = ex;
    __syncthreads();
    float sum = exps[b][0] + exps[b][1] + exps[b][2] + exps[b][3];
    gates[b * EE + e] = ex / sum;
}

// ---------------------------------------------------------------------------
// 3) Convert x (f32) -> bf16, 4 elements per thread.
// ---------------------------------------------------------------------------
__global__ __launch_bounds__(256) void cvt_kernel(const float* __restrict__ x,
                                                  __bf16* __restrict__ xb) {
    const size_t i = ((size_t)blockIdx.x * 256 + threadIdx.x) * 4;
    float4 f = *(const float4*)(x + i);
    bf16x4 o;
    o.x = (__bf16)f.x; o.y = (__bf16)f.y; o.z = (__bf16)f.z; o.w = (__bf16)f.w;
    *(bf16x4*)(xb + i) = o;
}

// ---------------------------------------------------------------------------
// 4) Combine expert weights per sample (gate-weighted sum), emit bf16 in the
//    GEMM layout: wbuf[b][o][kpos*256 + c],  kpos = kh*3+kw.
//    W_experts layout: [E][O][C][3][3] f32.
// ---------------------------------------------------------------------------
__global__ __launch_bounds__(256) void combine_kernel(const float* __restrict__ We,
                                                      const float* __restrict__ gates,
                                                      __bf16* __restrict__ wbuf) {
    const size_t idx = (size_t)blockIdx.x * 256 + threadIdx.x;   // < B*O*KTOT
    const int k    = (int)(idx % KTOT);
    const int o    = (int)((idx / KTOT) % OO);
    const int b    = (int)(idx / ((size_t)KTOT * OO));
    const int kpos = k >> 8;       // 0..8
    const int c    = k & 255;
    const size_t ebase = (((size_t)o * CC + c) * (KK * KK)) + kpos;
    const float* gp = gates + b * EE;
    float v = 0.f;
    #pragma unroll
    for (int e = 0; e < EE; ++e)
        v += gp[e] * We[(size_t)e * (OO * CC * KK * KK) + ebase];
    wbuf[idx] = (__bf16)v;
}

// ---------------------------------------------------------------------------
// Issue one thread's 64B weight-row copy (global -> LDS) as 4 async b128 DMAs.
// The instruction offset applies to both the LDS and global addresses.
// ---------------------------------------------------------------------------
__device__ __forceinline__ void issue_async_a(const __bf16* src, __bf16* lds_dst) {
    const unsigned lds = (unsigned)(size_t)lds_dst;   // low 32 bits = LDS offset
    asm volatile(
        "global_load_async_to_lds_b128 %0, %1, off\n\t"
        "global_load_async_to_lds_b128 %0, %1, off offset:16\n\t"
        "global_load_async_to_lds_b128 %0, %1, off offset:32\n\t"
        "global_load_async_to_lds_b128 %0, %1, off offset:48"
        :
        : "v"(lds), "v"(src)
        : "memory");
}

// ---------------------------------------------------------------------------
// 5) Implicit-GEMM 3x3 conv, bf16 WMMA, f32 accum.
//    Block: 256 threads (8 wave32). Output tile: M=256 (all out-chans) x
//    N=64 (one image row). Each wave owns a 32-row M strip: 2 A fragments,
//    8 accumulators -> 8 WMMAs per kpos step.
//    A tiles are fetched by the async DMA path (global_load_async_to_lds,
//    ASYNCcnt), pipelined one step deep into a TRIPLE-buffered LDS tile
//    (mod-3 rotation makes the prefetch target disjoint from both live
//    read buffers under a single barrier per step). 9 steps/slice and
//    9 % 3 == 0 keep the buffer index (kp % 3) compile-time constant.
// ---------------------------------------------------------------------------
__global__ __launch_bounds__(256) void moe_conv_kernel(const __bf16* __restrict__ xb,
                                                       const __bf16* __restrict__ wbuf,
                                                       float* __restrict__ out) {
    const int h0 = blockIdx.x;              // image row, 0..63  (N tile)
    const int b  = blockIdx.y;              // 0..15

    __shared__ __align__(16) __bf16 As[3][256][APITCH];  // 60.0 KB (triple buf)
    __shared__ __align__(16) __bf16 Xs[3][66][XPITCH];   // 15.5 KB (halo cols 0,65)

    const int tid  = threadIdx.x;
    const int wave = tid >> 5;
    const int lane = tid & 31;

    v8f acc[2][4] = {};

    // Zero the halo columns once (always zero for every slice).
    if (tid < 192) {
        const int r    = tid / 64;
        const int side = (tid >> 5) & 1;
        const int c    = tid & 31;
        Xs[r][side ? 65 : 0][c] = (__bf16)0.0f;
    }

    const size_t wbase = (size_t)b * OO * KTOT;

    // x-stage thread mapping: c = tid>>3 (0..31), pixel group g = tid&7
    const int xc = tid >> 3;
    const int xg = tid & 7;
    // A-stage thread mapping: one full 32-elem (64B) weight row per thread
    const __bf16* awrow = wbuf + wbase + (size_t)tid * KTOT;

    // Prologue: issue DMA for step 0 (cs=0, kp=0) into buffer 0.
    issue_async_a(awrow, &As[0][tid][0]);

    for (int cs = 0; cs < 8; ++cs) {
        const int c0 = cs << 5;
        __syncthreads();   // Xs WAR: previous slice fully consumed

        // ---- stage Xs: 32 channels x 3 rows x 64 px, transposed, padded ----
        {
            const size_t chbase = ((size_t)b * CC + c0 + xc) * (HH * WW);
            #pragma unroll
            for (int r = 0; r < 3; ++r) {
                const int hp = h0 + r - 1;
                Pack8 p; p.q = (u32x4){0u, 0u, 0u, 0u};
                if ((unsigned)hp < (unsigned)HH)
                    p.q = *(const u32x4*)(xb + chbase + (size_t)hp * WW + xg * 8);
                #pragma unroll
                for (int j = 0; j < 8; ++j)
                    Xs[r][xg * 8 + 1 + j][xc] = p.h[j];
            }
        }

        #pragma unroll
        for (int kp = 0; kp < 9; ++kp) {
            const int abuf = kp % 3;                  // (cs*9+kp)%3, 9%3==0
            const int nbuf = (kp + 1) % 3;

            // ---- issue next step's A-tile DMA; wait for current tile ----
            if (kp < 8) {
                issue_async_a(awrow + (kp + 1) * 256 + c0, &As[nbuf][tid][0]);
                asm volatile("s_wait_asynccnt 0x4" ::: "memory");
            } else if (cs < 7) {
                issue_async_a(awrow + (c0 + 32), &As[nbuf][tid][0]);
                asm volatile("s_wait_asynccnt 0x4" ::: "memory");
            } else {
                asm volatile("s_wait_asynccnt 0x0" ::: "memory");
            }
            __syncthreads();  // A[abuf] landed in all waves (Xs ready at kp==0)

            // ---- A fragments (2 M sub-blocks of the wave's 32-row strip):
            //      lanes 0-15 row M=lane, K {0..7,16..23};
            //      lanes 16-31 row M=lane-16, K {8..15,24..31} ----
            const int arow  = (wave << 5) + (lane & 15);
            const int akoff = (lane >> 4) << 3;       // 0 or 8
            Frag a0, a1;
            a0.q[0] = *(const u32x4*)&As[abuf][arow][akoff];
            a0.q[1] = *(const u32x4*)&As[abuf][arow][akoff + 16];
            a1.q[0] = *(const u32x4*)&As[abuf][arow + 16][akoff];
            a1.q[1] = *(const u32x4*)&As[abuf][arow + 16][akoff + 16];

            // ---- B fragments from Xs: lane<16 col N=lane, K(ch) {0..7,16..23};
            //      lane>=16 col N=lane-16, K {8..15,24..31}.  Pixel = N + kw (halo+1).
            const int kh = kp / 3;
            const int kw = kp % 3;
            const int bk = (lane >> 4) << 3;          // 0 or 8
            #pragma unroll
            for (int nb = 0; nb < 4; ++nb) {
                const int col = nb * 16 + (lane & 15) + kw;
                const __bf16* bp = &Xs[kh][col][bk];
                Frag bf;
                bf.q[0] = *(const u32x4*)bp;
                bf.q[1] = *(const u32x4*)(bp + 16);
                acc[0][nb] = __builtin_amdgcn_wmma_f32_16x16x32_bf16(
                    false, a0.v, false, bf.v, (short)0, acc[0][nb], false, false);
                acc[1][nb] = __builtin_amdgcn_wmma_f32_16x16x32_bf16(
                    false, a1.v, false, bf.v, (short)0, acc[1][nb], false, false);
            }
        }
    }

    // ---- epilogue: f32 C/D layout -> out[b][o][h0][w] ----
    // VGPR r: lanes 0-15 -> (M=r, N=lane); lanes 16-31 -> (M=r+8, N=lane-16)
    const int obase = (wave << 5) + ((lane >> 4) << 3);
    const int npix0 = lane & 15;
    #pragma unroll
    for (int mi = 0; mi < 2; ++mi) {
        #pragma unroll
        for (int nb = 0; nb < 4; ++nb) {
            const int wpix = nb * 16 + npix0;
            #pragma unroll
            for (int r = 0; r < 8; ++r) {
                const int o = obase + mi * 16 + r;
                out[(((size_t)b * OO + o) * HH + h0) * WW + wpix] = acc[mi][nb][r];
            }
        }
    }
}

// ---------------------------------------------------------------------------
extern "C" void kernel_launch(void* const* d_in, const int* in_sizes, int n_in,
                              void* d_out, int out_size, void* d_ws, size_t ws_size,
                              hipStream_t stream) {
    const float* x      = (const float*)d_in[0];   // [B,C,H,W]
    const float* We     = (const float*)d_in[1];   // [E,O,C,3,3]
    const float* gate_w = (const float*)d_in[2];   // [E,C]
    float* out          = (float*)d_out;           // [B,O,H,W]

    // Workspace layout
    char* ws = (char*)d_ws;
    __bf16* xb     = (__bf16*)ws;                                 // 33,554,432 B
    __bf16* wbuf   = (__bf16*)(ws + (size_t)33554432);            // 18,874,368 B
    float*  pooled = (float*)(ws + (size_t)33554432 + 18874368);  // 16 KB
    float*  gates  = pooled + BB * CC;                            // 256 B

    // 1) pooled
    pool_kernel<<<BB * CC, 256, 0, stream>>>(x, pooled);
    // 2) gates
    gate_kernel<<<1, 64, 0, stream>>>(pooled, gate_w, gates);
    // 3) x -> bf16
    {
        const size_t n   = (size_t)BB * CC * HH * WW;   // 16,777,216
        const int blocks = (int)(n / (256 * 4));        // 16384
        cvt_kernel<<<blocks, 256, 0, stream>>>(x, xb);
    }
    // 4) combined per-sample weights (bf16, GEMM layout)
    {
        const size_t n   = (size_t)BB * OO * KTOT;      // 9,437,184
        const int blocks = (int)(n / 256);              // 36864
        combine_kernel<<<blocks, 256, 0, stream>>>(We, gates, wbuf);
    }
    // 5) WMMA implicit-GEMM conv (async-DMA A tiles)
    {
        dim3 grid(HH, BB);                              // (64, 16)
        moe_conv_kernel<<<grid, 256, 0, stream>>>(xb, wbuf, out);
    }
}
